// GNN_Location_60052232732947
// MI455X (gfx1250) — matrix operand order repaired and checked by reading, compiled, not run on hardware
//
#include <hip/hip_runtime.h>
#include <hip/hip_fp16.h>

// ---------------------------------------------------------------------------
// GNN earthquake-location forward for MI455X (gfx1250, wave32, WMMA).
//
// Heavy op: 20x edge-message GEMM [E=1.2M x 33] @ [33 x 30] -> WMMA f16,
// staged via GLOBAL_LOAD_ASYNC_TO_LDS_B128 (ASYNCcnt).  Working set ~70MB
// fits in 192MB L2 -> L2-resident; scatter-mean via native f32 L2 atomics
// (sum buffer padded to 32 cols so the scatter is branch-free).  Edge-degree
// counts are loop-invariant -> computed once.  Node MLPs are skinny -> VALU
// with LDS-staged weights.
// ---------------------------------------------------------------------------

#define CDIV(a, b) (((a) + (b) - 1) / (b))
#define BLOCK 256
#define WPB 8 // waves per block (wave32)

typedef __attribute__((ext_vector_type(16))) _Float16 v16h_t;
typedef __attribute__((ext_vector_type(8)))  _Float16 v8h_t;
typedef __attribute__((ext_vector_type(8)))  float    v8f_t;
typedef __attribute__((ext_vector_type(4)))  int      v4i_t;

typedef __attribute__((address_space(1))) v4i_t gv4i_t; // global int4
typedef __attribute__((address_space(3))) v4i_t lv4i_t; // LDS int4

__device__ __forceinline__ float prelu_f(float x) { return x >= 0.f ? x : 0.25f * x; }
__device__ __forceinline__ void atomAddF(float* p, float v) { unsafeAtomicAdd(p, v); }

// async global->LDS copy, 16B per lane (GLOBAL_LOAD_ASYNC_TO_LDS_B128)
__device__ __forceinline__ void async_g2l_b128(const void* g, void* l) {
#if defined(__has_builtin) && __has_builtin(__builtin_amdgcn_global_load_async_to_lds_b128)
    __builtin_amdgcn_global_load_async_to_lds_b128((gv4i_t*)g, (lv4i_t*)l, 0, 0);
#else
    asm volatile("global_load_async_to_lds_b128 %0, %1, off"
                 :: "v"((unsigned)(unsigned long long)l), "v"(g) : "memory");
#endif
}
__device__ __forceinline__ void wait_async0() {
#if defined(__has_builtin) && __has_builtin(__builtin_amdgcn_s_wait_asynccnt)
    __builtin_amdgcn_s_wait_asynccnt(0);
    asm volatile("" ::: "memory");
#else
    asm volatile("s_wait_asynccnt 0" ::: "memory");
#endif
}

// ---------------------------------------------------------------------------
__global__ void k_zero(float* __restrict__ p, long long n) {
    long long i = (long long)blockIdx.x * blockDim.x + threadIdx.x;
    if (i < n) p[i] = 0.f;
}

// per-destination edge degree (loop-invariant across all props)
__global__ void k_count(const int* __restrict__ e1, float* __restrict__ cnt, int E) {
    int e = blockIdx.x * blockDim.x + threadIdx.x;
    if (e < E) atomAddF(&cnt[e1[e]], 1.0f);
}

// relh[e][0..2] = (coords[map[e0]] - coords[map[e1]]) * scale  (f16, pad to 4)
__global__ void k_rel(const int* __restrict__ A, const int* __restrict__ map,
                      const float* __restrict__ coords, _Float16* __restrict__ relh,
                      int E, float scale) {
    int e = blockIdx.x * blockDim.x + threadIdx.x;
    if (e >= E) return;
    int i0 = map[A[e]], i1 = map[A[E + e]];
    #pragma unroll
    for (int k = 0; k < 3; ++k)
        relh[e * 4 + k] = (_Float16)((coords[i0 * 3 + k] - coords[i1 * 3 + k]) * scale);
    relh[e * 4 + 3] = (_Float16)0.f;
}

// ---------------------------------------------------------------------------
// Pack merge weight W[30,33] into 4 WMMA B fragments (f16).
// Edge-feature K layout: K=0..29 -> x, K=30,31 -> 0, K=32..34 -> rel, rest 0.
// B 32x16 layout: lanes 0-15 col N=lane, K=0..15 across the 16 halves;
// lanes 16-31 col N=lane-16, K=16..31.
__global__ void k_pack_w(const float* __restrict__ W, _Float16* __restrict__ bpack) {
    int frag = threadIdx.x >> 5, lane = threadIdx.x & 31;
    int nt = frag >> 1, kt = frag & 1;
    #pragma unroll
    for (int h = 0; h < 16; ++h) {
        int K = ((lane >> 4) << 4) + h + (kt << 5);
        int Ncol = (lane & 15) + (nt << 4);
        float v = 0.f;
        if (Ncol < 30) {
            if (K < 30) v = W[Ncol * 33 + K];
            else if (K >= 32 && K < 35) v = W[Ncol * 33 + 30 + (K - 32)];
        }
        bpack[(frag * 32 + lane) * 16 + h] = (_Float16)v;
    }
}

// ---------------------------------------------------------------------------
// Core edge-propagation: per 16-edge tile, async-gather f16 rows into LDS,
// WMMA 16x16x32_f16 x4 (K padded 35->64, N padded 30->32), PReLU, scatter.
// Scatter is branch-free: cols 30,31 of `sum` are write-only padding.
__global__ void k_prop_wmma(const _Float16* __restrict__ xh,   // [N,32] f16 (cols 30,31 = 0)
                            const _Float16* __restrict__ relh, // [E,4] f16
                            const int* __restrict__ e0, const int* __restrict__ e1,
                            const _Float16* __restrict__ bpack, // [4][32][16] f16
                            const float* __restrict__ bias,     // [30]
                            float* __restrict__ sum, int E) {
    __shared__ __align__(16) _Float16 tile[WPB][16][64];
    const int lane = threadIdx.x & 31;
    const int wave = threadIdx.x >> 5;
    const int gwaves = gridDim.x * WPB;
    const int ntiles = (E + 15) >> 4;

    v16h_t bf[2][2];
    #pragma unroll
    for (int nt = 0; nt < 2; ++nt)
        #pragma unroll
        for (int kt = 0; kt < 2; ++kt)
            bf[nt][kt] = *(const v16h_t*)(bpack + (((nt << 1) | kt) * 32 + lane) * 16);

    const int col = lane & 15;
    const float b0 = bias[col];
    const float b1 = (col + 16 < 30) ? bias[col + 16] : 0.f; // padded cols get 0 bias

    for (int t = blockIdx.x * WPB + wave; t < ntiles; t += gwaves) {
        const int base = t << 4;

        // ---- async-stage x rows (K halves 0..31) : 2 lanes per row, 32B each
        {
            const int arow = lane >> 1;
            const int achk = lane & 1;
            const int ae = base + arow;
            const int aec = (ae < E) ? ae : (E - 1);
            const char* g = (const char*)(xh + (size_t)e0[aec] * 32) + achk * 32;
            char* l = (char*)&tile[wave][arow][0] + achk * 32;
            async_g2l_b128(g, l);
            async_g2l_b128(g + 16, l + 16);
        }
        // ---- DS-stage upper K half: zeros + rel at K=32..34 (disjoint from async)
        if (lane >= 16) {
            const int m = lane & 15;
            const int e = base + m;
            const int ec = (e < E) ? e : (E - 1);
            v8h_t z = {};
            v8h_t* dst = (v8h_t*)&tile[wave][m][32];
            dst[0] = z; dst[1] = z; dst[2] = z; dst[3] = z;
            tile[wave][m][32] = relh[(size_t)ec * 4 + 0];
            tile[wave][m][33] = relh[(size_t)ec * 4 + 1];
            tile[wave][m][34] = relh[(size_t)ec * 4 + 2];
        }
        {
            int tn = t + gwaves;
            if (tn < ntiles) __builtin_prefetch(e0 + (tn << 4), 0, 1);
        }
        wait_async0();                                   // x rows in LDS
        asm volatile("s_wait_dscnt 0" ::: "memory");     // rel/zero stores in LDS

        // A fragment, 16x32, in contiguous K order (not the ISA table order):
        // lanes 0-15: K 0-15 ; lanes 16-31: K 16-31 (contiguous, matches B pack)
        const int row = lane & 15;
        const int hi8 = (lane >> 4) << 4;
        const _Float16* tp = &tile[wave][row][0];
        union { v16h_t v; struct { v8h_t lo, hi; } p; } a0, a1;
        a0.p.lo = *(const v8h_t*)(tp + hi8);
        a0.p.hi = *(const v8h_t*)(tp + 8 + hi8);
        a1.p.lo = *(const v8h_t*)(tp + 32 + hi8);
        a1.p.hi = *(const v8h_t*)(tp + 40 + hi8);

        v8f_t c0 = {}, c1 = {};
        c0 = __builtin_amdgcn_wmma_f32_16x16x32_f16(false, a0.v, false, bf[0][0], (short)0, c0, false, false);
        c0 = __builtin_amdgcn_wmma_f32_16x16x32_f16(false, a1.v, false, bf[0][1], (short)0, c0, false, false);
        c1 = __builtin_amdgcn_wmma_f32_16x16x32_f16(false, a0.v, false, bf[1][0], (short)0, c1, false, false);
        c1 = __builtin_amdgcn_wmma_f32_16x16x32_f16(false, a1.v, false, bf[1][1], (short)0, c1, false, false);

        // D layout: lane holds col N=(lane&15); VGPR r -> row M = r + 8*(lane>=16)
        const int rbase = (lane >> 4) << 3;
        if (base + 16 <= E) {
            // full tile: branch-free scatter; dst indices via 2x b128 loads
            v4i_t d03 = *(const v4i_t*)(e1 + base + rbase);
            v4i_t d47 = *(const v4i_t*)(e1 + base + rbase + 4);
            #pragma unroll
            for (int r = 0; r < 8; ++r) {
                const int dst = (r < 4) ? d03[r & 3] : d47[r & 3];
                float* sp = &sum[(size_t)dst * 32 + col];
                atomAddF(sp, prelu_f(c0[r] + b0));
                atomAddF(sp + 16, prelu_f(c1[r] + b1));
            }
        } else {
            #pragma unroll
            for (int r = 0; r < 8; ++r) {
                const int ee = base + rbase + r;
                if (ee < E) {
                    const int dst = e1[ee];
                    float* sp = &sum[(size_t)dst * 32 + col];
                    atomAddF(sp, prelu_f(c0[r] + b0));
                    atomAddF(sp + 16, prelu_f(c1[r] + b1));
                }
            }
        }
    }
}

// ---------------------------------------------------------------------------
// mean = sum / max(cnt,1), in place over [n,32]
__global__ void k_div(float* __restrict__ sum, const float* __restrict__ cnt, int n) {
    long long i = (long long)blockIdx.x * blockDim.x + threadIdx.x;
    if (i >= (long long)n * 32) return;
    sum[i] /= fmaxf(cnt[i >> 5], 1.f);
}

// ---------------------------------------------------------------------------
// mask embed: 18 -> 20 (PReLU) -> 10, no output activation
__global__ void k_embed(const float* __restrict__ mraw,
                        const float* __restrict__ W0, const float* __restrict__ B0,
                        const float* __restrict__ W1, const float* __restrict__ B1,
                        float* __restrict__ maske, int n) {
    __shared__ float sW0[20 * 18], sb0[20], sW1[10 * 20], sb1[10];
    for (int i = threadIdx.x; i < 20 * 18; i += blockDim.x) sW0[i] = W0[i];
    for (int i = threadIdx.x; i < 10 * 20; i += blockDim.x) sW1[i] = W1[i];
    if (threadIdx.x < 20) sb0[threadIdx.x] = B0[threadIdx.x];
    if (threadIdx.x < 10) sb1[threadIdx.x] = B1[threadIdx.x];
    __syncthreads();
    int nd = blockIdx.x * blockDim.x + threadIdx.x;
    if (nd >= n) return;
    float x[18], t[20];
    for (int i = 0; i < 18; ++i) x[i] = mraw[(size_t)nd * 18 + i];
    for (int o = 0; o < 20; ++o) {
        float a = sb0[o];
        for (int i = 0; i < 18; ++i) a += sW0[o * 18 + i] * x[i];
        t[o] = prelu_f(a);
    }
    for (int o = 0; o < 10; ++o) {
        float a = sb1[o];
        for (int i = 0; i < 20; ++i) a += sW1[o * 20 + i] * t[i];
        maske[(size_t)nd * 16 + o] = a;
    }
}

// ---------------------------------------------------------------------------
// init: tr30 = PReLU(W [cin+10 -> 30] @ [h ; mask]); xh = f16(PReLU(tr30))
__global__ void k_init(const float* __restrict__ hin, int hstride, int cin,
                       const float* __restrict__ maske,
                       const float* __restrict__ W, const float* __restrict__ B,
                       float* __restrict__ tr30, _Float16* __restrict__ xh, int n) {
    __shared__ float sW[30 * 40], sb[30];
    const int din = cin + 10;
    for (int i = threadIdx.x; i < 30 * din; i += blockDim.x) sW[i] = W[i];
    if (threadIdx.x < 30) sb[threadIdx.x] = B[threadIdx.x];
    __syncthreads();
    int nd = blockIdx.x * blockDim.x + threadIdx.x;
    if (nd >= n) return;
    float x[40];
    for (int i = 0; i < cin; ++i) x[i] = hin[(size_t)nd * hstride + i];
    for (int i = 0; i < 10; ++i) x[cin + i] = maske[(size_t)nd * 16 + i];
    for (int o = 0; o < 30; ++o) {
        float a = sb[o];
        for (int i = 0; i < din; ++i) a += sW[o * din + i] * x[i];
        float t = prelu_f(a);
        tr30[(size_t)nd * 32 + o] = t;
        xh[(size_t)nd * 32 + o] = (_Float16)prelu_f(t);
    }
    xh[(size_t)nd * 32 + 30] = (_Float16)0.f;
    xh[(size_t)nd * 32 + 31] = (_Float16)0.f;
}

// ---------------------------------------------------------------------------
// l1: tr60 = PReLU(concat[l1a(tr,msta,mask), l1b(tr,msrc,mask)])  (70->30 x2)
__global__ void k_l1(const float* __restrict__ tr30, const float* __restrict__ msta,
                     const float* __restrict__ msrc, const float* __restrict__ maske,
                     const float* __restrict__ Wa, const float* __restrict__ Ba,
                     const float* __restrict__ Wb, const float* __restrict__ Bb,
                     float* __restrict__ tr60, int n) {
    __shared__ float sWa[30 * 70], sWb[30 * 70], sba[30], sbb[30];
    for (int i = threadIdx.x; i < 30 * 70; i += blockDim.x) { sWa[i] = Wa[i]; sWb[i] = Wb[i]; }
    if (threadIdx.x < 30) { sba[threadIdx.x] = Ba[threadIdx.x]; sbb[threadIdx.x] = Bb[threadIdx.x]; }
    __syncthreads();
    int nd = blockIdx.x * blockDim.x + threadIdx.x;
    if (nd >= n) return;
    float xa[70], xb[70];
    for (int i = 0; i < 30; ++i) { float t = tr30[(size_t)nd * 32 + i]; xa[i] = t; xb[i] = t; }
    for (int i = 0; i < 30; ++i) { xa[30 + i] = msta[(size_t)nd * 32 + i]; xb[30 + i] = msrc[(size_t)nd * 32 + i]; }
    for (int i = 0; i < 10; ++i) { float m = maske[(size_t)nd * 16 + i]; xa[60 + i] = m; xb[60 + i] = m; }
    for (int o = 0; o < 30; ++o) {
        float a = sba[o], b = sbb[o];
        for (int i = 0; i < 70; ++i) { a += sWa[o * 70 + i] * xa[i]; b += sWb[o * 70 + i] * xb[i]; }
        tr60[(size_t)nd * 64 + o] = prelu_f(a);
        tr60[(size_t)nd * 64 + 30 + o] = prelu_f(b);
    }
}

// ---------------------------------------------------------------------------
// xh = f16(PReLU(W [60->30] @ tr60 + b))  (for second-round props)
__global__ void k_linh(const float* __restrict__ tr60, const float* __restrict__ W,
                       const float* __restrict__ B, _Float16* __restrict__ xh, int n) {
    __shared__ float sW[30 * 60], sb[30];
    for (int i = threadIdx.x; i < 30 * 60; i += blockDim.x) sW[i] = W[i];
    if (threadIdx.x < 30) sb[threadIdx.x] = B[threadIdx.x];
    __syncthreads();
    int nd = blockIdx.x * blockDim.x + threadIdx.x;
    if (nd >= n) return;
    float x[60];
    for (int i = 0; i < 60; ++i) x[i] = tr60[(size_t)nd * 64 + i];
    for (int o = 0; o < 30; ++o) {
        float a = sb[o];
        for (int i = 0; i < 60; ++i) a += sW[o * 60 + i] * x[i];
        xh[(size_t)nd * 32 + o] = (_Float16)prelu_f(a);
    }
    xh[(size_t)nd * 32 + 30] = (_Float16)0.f;
    xh[(size_t)nd * 32 + 31] = (_Float16)0.f;
}

// ---------------------------------------------------------------------------
// l2: h = PReLU(concat[l2a2(tr60,msta,mask), l2b2(tr60,msrc,mask)]) (100->15 x2)
__global__ void k_l2(const float* __restrict__ tr60, const float* __restrict__ msta,
                     const float* __restrict__ msrc, const float* __restrict__ maske,
                     const float* __restrict__ Wa, const float* __restrict__ Ba,
                     const float* __restrict__ Wb, const float* __restrict__ Bb,
                     float* __restrict__ h, int n) {
    __shared__ float sWa[15 * 100], sWb[15 * 100], sba[15], sbb[15];
    for (int i = threadIdx.x; i < 15 * 100; i += blockDim.x) { sWa[i] = Wa[i]; sWb[i] = Wb[i]; }
    if (threadIdx.x < 15) { sba[threadIdx.x] = Ba[threadIdx.x]; sbb[threadIdx.x] = Bb[threadIdx.x]; }
    __syncthreads();
    int nd = blockIdx.x * blockDim.x + threadIdx.x;
    if (nd >= n) return;
    float x60[60], ma[30], mb[30], mk[10];
    for (int i = 0; i < 60; ++i) x60[i] = tr60[(size_t)nd * 64 + i];
    for (int i = 0; i < 30; ++i) { ma[i] = msta[(size_t)nd * 32 + i]; mb[i] = msrc[(size_t)nd * 32 + i]; }
    for (int i = 0; i < 10; ++i) mk[i] = maske[(size_t)nd * 16 + i];
    for (int o = 0; o < 15; ++o) {
        float a = sba[o], b = sbb[o];
        for (int i = 0; i < 60; ++i) { a += sWa[o * 100 + i] * x60[i]; b += sWb[o * 100 + i] * x60[i]; }
        for (int i = 0; i < 30; ++i) { a += sWa[o * 100 + 60 + i] * ma[i]; b += sWb[o * 100 + 60 + i] * mb[i]; }
        for (int i = 0; i < 10; ++i) { a += sWa[o * 100 + 90 + i] * mk[i]; b += sWb[o * 100 + 90 + i] * mk[i]; }
        h[(size_t)nd * 32 + o] = prelu_f(a);
        h[(size_t)nd * 32 + 15 + o] = prelu_f(b);
    }
}

// ---------------------------------------------------------------------------
// readout edge stage: per pick node, fc1a(43->30) PReLU, fc1b(30->30) PReLU,
// scatter-mean to its source.
__global__ void k_ro(const float* __restrict__ h, const float* __restrict__ maske,
                     const int* __restrict__ sta_of, const int* __restrict__ src_of,
                     const float* __restrict__ locs, const float* __restrict__ srcc,
                     const float* __restrict__ W1, const float* __restrict__ B1,
                     const float* __restrict__ W2, const float* __restrict__ B2,
                     float* __restrict__ sum, float* __restrict__ cnt, int n) {
    __shared__ float sW1[30 * 43], sB1[30], sW2[30 * 30], sB2[30];
    for (int i = threadIdx.x; i < 30 * 43; i += blockDim.x) sW1[i] = W1[i];
    for (int i = threadIdx.x; i < 30 * 30; i += blockDim.x) sW2[i] = W2[i];
    if (threadIdx.x < 30) { sB1[threadIdx.x] = B1[threadIdx.x]; sB2[threadIdx.x] = B2[threadIdx.x]; }
    __syncthreads();
    int nd = blockIdx.x * blockDim.x + threadIdx.x;
    if (nd >= n) return;
    int st = sta_of[nd], sc = src_of[nd];
    float x[43], t[30];
    for (int i = 0; i < 30; ++i) x[i] = h[(size_t)nd * 32 + i];
    for (int i = 0; i < 10; ++i) x[30 + i] = maske[(size_t)nd * 16 + i];
    for (int k = 0; k < 3; ++k) x[40 + k] = (srcc[sc * 3 + k] - locs[st * 3 + k]) * 1e-4f;
    for (int o = 0; o < 30; ++o) {
        float a = sB1[o];
        for (int i = 0; i < 43; ++i) a += sW1[o * 43 + i] * x[i];
        t[o] = prelu_f(a);
    }
    for (int o = 0; o < 30; ++o) {
        float a = sB2[o];
        for (int i = 0; i < 30; ++i) a += sW2[o * 30 + i] * t[i];
        atomAddF(&sum[(size_t)sc * 32 + o], prelu_f(a));
    }
    atomAddF(&cnt[sc], 1.0f);
}

// ---------------------------------------------------------------------------
// per-source: mean, fc2(30->15) PReLU, proj(15->30->3) & proj_t(15->15->1)
__global__ void k_out(const float* __restrict__ sum, const float* __restrict__ cnt,
                      const float* __restrict__ W2, const float* __restrict__ B2,
                      const float* __restrict__ P0W, const float* __restrict__ P0B,
                      const float* __restrict__ P1W, const float* __restrict__ P1B,
                      const float* __restrict__ T0W, const float* __restrict__ T0B,
                      const float* __restrict__ T1W, const float* __restrict__ T1B,
                      float* __restrict__ out, int nsrc) {
    int s = blockIdx.x * blockDim.x + threadIdx.x;
    if (s >= nsrc) return;
    float c = fmaxf(cnt[s], 1.f);
    float m[30], hs[15], t30[30], t15[15];
    for (int i = 0; i < 30; ++i) m[i] = sum[(size_t)s * 32 + i] / c;
    for (int o = 0; o < 15; ++o) {
        float a = B2[o];
        for (int i = 0; i < 30; ++i) a += W2[o * 30 + i] * m[i];
        hs[o] = prelu_f(a);
    }
    for (int o = 0; o < 30; ++o) {
        float a = P0B[o];
        for (int i = 0; i < 15; ++i) a += P0W[o * 15 + i] * hs[i];
        t30[o] = prelu_f(a);
    }
    for (int o = 0; o < 3; ++o) {
        float a = P1B[o];
        for (int i = 0; i < 30; ++i) a += P1W[o * 30 + i] * t30[i];
        out[(size_t)s * 4 + o] = a;
    }
    for (int o = 0; o < 15; ++o) {
        float a = T0B[o];
        for (int i = 0; i < 15; ++i) a += T0W[o * 15 + i] * hs[i];
        t15[o] = prelu_f(a);
    }
    {
        float a = T1B[0];
        for (int i = 0; i < 15; ++i) a += T1W[i] * t15[i];
        out[(size_t)s * 4 + 3] = a;
    }
}

// ---------------------------------------------------------------------------
extern "C" void kernel_launch(void* const* d_in, const int* in_sizes, int n_in,
                              void* d_out, int out_size, void* d_ws, size_t ws_size,
                              hipStream_t stream) {
    (void)n_in; (void)out_size; (void)ws_size;
    const float* tr_in    = (const float*)d_in[0];
    const float* mask_raw = (const float*)d_in[1];
    const int*   A_sta    = (const int*)d_in[2];
    const int*   A_src    = (const int*)d_in[3];
    const int*   A_map    = (const int*)d_in[4]; // [2,N]: row0 sta_of, row1 src_of
    const float* locs     = (const float*)d_in[6];
    const float* srcc     = (const float*)d_in[7];

    const int N  = in_sizes[0] / 18;
    const int E1 = in_sizes[2] / 2;
    const int E2 = in_sizes[3] / 2;
    const int NS = 5000;
    const int* sta_of = A_map;
    const int* src_of = A_map + N;

    // ---- parameter walker (leaves in dict insertion order) ----
    int pi = 8;
    struct Lin { const float* W; const float* b; };
    auto next = [&]() { Lin l{(const float*)d_in[pi], (const float*)d_in[pi + 1]}; pi += 2; return l; };
    Lin emb0 = next(), emb1 = next();
    Lin da[5][8]; // 0 init,1 l1a,2 l1b,3 l2a1,4 l2a2,5 l2b1,6 l2b2,7 merge
    for (int l = 0; l < 5; ++l)
        for (int k = 0; k < 8; ++k) da[l][k] = next();
    Lin fc1a = next(), fc1b = next(), fc2 = next();
    Lin proj0 = next(), proj1 = next(), pt0 = next(), pt1 = next();

    // ---- workspace layout ----
    char* wp = (char*)d_ws;
    auto alloc = [&](size_t bytes) { void* p = wp; wp += (bytes + 255) & ~(size_t)255; return p; };
    _Float16* relh_sta = (_Float16*)alloc((size_t)E1 * 4 * 2);
    _Float16* relh_src = (_Float16*)alloc((size_t)E2 * 4 * 2);
    _Float16* bpack    = (_Float16*)alloc((size_t)5 * 4 * 32 * 16 * 2);
    float*    maske    = (float*)alloc((size_t)N * 16 * 4);
    float*    hbuf     = (float*)alloc((size_t)N * 32 * 4);
    float*    tr30     = (float*)alloc((size_t)N * 32 * 4);
    float*    tr60     = (float*)alloc((size_t)N * 64 * 4);
    _Float16* xh       = (_Float16*)alloc((size_t)N * 32 * 2);
    float*    sum_sta  = (float*)alloc((size_t)N * 32 * 4);
    float*    cnt_sta  = (float*)alloc((size_t)N * 4);
    float*    sum_src  = (float*)alloc((size_t)N * 32 * 4);
    float*    cnt_src  = (float*)alloc((size_t)N * 4);
    float*    sum_ro   = (float*)alloc((size_t)NS * 32 * 4);
    float*    cnt_ro   = (float*)alloc((size_t)NS * 4);

    const int gN   = CDIV(N, BLOCK);
    const int gP1  = CDIV(CDIV(E1, 16), WPB);
    const int gP2  = CDIV(CDIV(E2, 16), WPB);
    const int gDiv = CDIV(N * 32, BLOCK);

    // ---- precompute (rel vectors, weight packs, degrees, mask embed) ----
    k_rel<<<CDIV(E1, BLOCK), BLOCK, 0, stream>>>(A_sta, sta_of, locs, relh_sta, E1, 1e-4f);
    k_rel<<<CDIV(E2, BLOCK), BLOCK, 0, stream>>>(A_src, src_of, srcc, relh_src, E2, 1e-4f);
    for (int l = 0; l < 5; ++l)
        k_pack_w<<<1, 128, 0, stream>>>(da[l][7].W, bpack + (size_t)l * 4 * 32 * 16);
    k_zero<<<gN, BLOCK, 0, stream>>>(cnt_sta, N);
    k_zero<<<gN, BLOCK, 0, stream>>>(cnt_src, N);
    k_count<<<CDIV(E1, BLOCK), BLOCK, 0, stream>>>(A_sta + E1, cnt_sta, E1);
    k_count<<<CDIV(E2, BLOCK), BLOCK, 0, stream>>>(A_src + E2, cnt_src, E2);
    k_embed<<<gN, BLOCK, 0, stream>>>(mask_raw, emb0.W, emb0.b, emb1.W, emb1.b, maske, N);

    // ---- 5 DataAggregation layers ----
    const float* hin = tr_in;
    int hstride = 18, cin = 18;
    for (int l = 0; l < 5; ++l) {
        const _Float16* bp = bpack + (size_t)l * 4 * 32 * 16;
        k_init<<<gN, BLOCK, 0, stream>>>(hin, hstride, cin, maske, da[l][0].W, da[l][0].b, tr30, xh, N);

        k_zero<<<gDiv, BLOCK, 0, stream>>>(sum_sta, (long long)N * 32);
        k_prop_wmma<<<gP1, BLOCK, 0, stream>>>(xh, relh_sta, A_sta, A_sta + E1, bp, da[l][7].b, sum_sta, E1);
        k_div<<<gDiv, BLOCK, 0, stream>>>(sum_sta, cnt_sta, N);

        k_zero<<<gDiv, BLOCK, 0, stream>>>(sum_src, (long long)N * 32);
        k_prop_wmma<<<gP2, BLOCK, 0, stream>>>(xh, relh_src, A_src, A_src + E2, bp, da[l][7].b, sum_src, E2);
        k_div<<<gDiv, BLOCK, 0, stream>>>(sum_src, cnt_src, N);

        k_l1<<<gN, BLOCK, 0, stream>>>(tr30, sum_sta, sum_src, maske,
                                       da[l][1].W, da[l][1].b, da[l][2].W, da[l][2].b, tr60, N);

        k_linh<<<gN, BLOCK, 0, stream>>>(tr60, da[l][3].W, da[l][3].b, xh, N);
        k_zero<<<gDiv, BLOCK, 0, stream>>>(sum_sta, (long long)N * 32);
        k_prop_wmma<<<gP1, BLOCK, 0, stream>>>(xh, relh_sta, A_sta, A_sta + E1, bp, da[l][7].b, sum_sta, E1);
        k_div<<<gDiv, BLOCK, 0, stream>>>(sum_sta, cnt_sta, N);

        k_linh<<<gN, BLOCK, 0, stream>>>(tr60, da[l][5].W, da[l][5].b, xh, N);
        k_zero<<<gDiv, BLOCK, 0, stream>>>(sum_src, (long long)N * 32);
        k_prop_wmma<<<gP2, BLOCK, 0, stream>>>(xh, relh_src, A_src, A_src + E2, bp, da[l][7].b, sum_src, E2);
        k_div<<<gDiv, BLOCK, 0, stream>>>(sum_src, cnt_src, N);

        k_l2<<<gN, BLOCK, 0, stream>>>(tr60, sum_sta, sum_src, maske,
                                       da[l][4].W, da[l][4].b, da[l][6].W, da[l][6].b, hbuf, N);
        hin = hbuf; hstride = 32; cin = 30;
    }

    // ---- bipartite readout ----
    k_zero<<<CDIV(NS * 32, BLOCK), BLOCK, 0, stream>>>(sum_ro, (long long)NS * 32);
    k_zero<<<CDIV(NS, BLOCK), BLOCK, 0, stream>>>(cnt_ro, NS);
    k_ro<<<gN, BLOCK, 0, stream>>>(hbuf, maske, sta_of, src_of, locs, srcc,
                                   fc1a.W, fc1a.b, fc1b.W, fc1b.b, sum_ro, cnt_ro, N);
    k_out<<<CDIV(NS, BLOCK), BLOCK, 0, stream>>>(sum_ro, cnt_ro, fc2.W, fc2.b,
                                                 proj0.W, proj0.b, proj1.W, proj1.b,
                                                 pt0.W, pt0.b, pt1.W, pt1.b,
                                                 (float*)d_out, NS);
}
